// LSTMNet_10462540333526
// MI455X (gfx1250) — matrix operand order repaired and checked by reading, not compiled, not run on hardware
//
#include <hip/hip_runtime.h>
#include <hip/hip_bf16.h>

// ---------------- problem constants (match reference) ----------------
#define BB   256      // batch
#define TT_  512      // timesteps
#define HH   512      // hidden
#define FUT_ 16
#define TTOT (TT_ + FUT_)   // 528
#define NWG  32       // workgroups: each owns 16 h-columns
#define BLK  512      // 16 wave32s: each wave owns one 16-row M tile

typedef _Float16 v16h __attribute__((ext_vector_type(16)));
typedef _Float16 v8h  __attribute__((ext_vector_type(8)));
typedef float    v8f  __attribute__((ext_vector_type(8)));

union AV { v16h v; v8h h[2]; };

__device__ __forceinline__ float sigf(float x) {
    return 1.0f / (1.0f + __expf(-x));
}

// -------- grid-wide barrier (arrive + generation), agent scope --------
__device__ __forceinline__ void gridbar(unsigned* bar, int tid) {
    __threadfence();            // make this thread's global writes visible device-wide
    __syncthreads();
    if (tid == 0) {
        unsigned g = __hip_atomic_load(&bar[1], __ATOMIC_RELAXED, __HIP_MEMORY_SCOPE_AGENT);
        unsigned a = __hip_atomic_fetch_add(&bar[0], 1u, __ATOMIC_ACQ_REL, __HIP_MEMORY_SCOPE_AGENT);
        if (a == (unsigned)(NWG - 1)) {
            __hip_atomic_store(&bar[0], 0u, __ATOMIC_RELAXED, __HIP_MEMORY_SCOPE_AGENT);
            __hip_atomic_fetch_add(&bar[1], 1u, __ATOMIC_RELEASE, __HIP_MEMORY_SCOPE_AGENT);
        } else {
            while (__hip_atomic_load(&bar[1], __ATOMIC_ACQUIRE, __HIP_MEMORY_SCOPE_AGENT) == g)
                __builtin_amdgcn_s_sleep(2);
        }
    }
    __syncthreads();
}

// -------- init: zero barrier state + parity-0 h buffers --------
__global__ void lstm_init(unsigned* bar, _Float16* h0, _Float16* h1) {
    int i = blockIdx.x * blockDim.x + threadIdx.x;
    if (i < 2) bar[i] = 0u;
    const int n = BB * HH;   // parity-0 only
    for (int j = i; j < n; j += gridDim.x * blockDim.x) {
        h0[j] = (_Float16)0.0f;
        h1[j] = (_Float16)0.0f;
    }
}

// load a 32-K-contiguous B tile (one gate) from LDS
__device__ __forceinline__ AV ldB(const _Float16* base) {
    AV b;
    b.h[0] = *(const v8h*)base;
    b.h[1] = *(const v8h*)(base + 8);
    return b;
}
// load an A tile (h rows) from global f16
__device__ __forceinline__ AV ldA(const _Float16* base) {
    AV a;
    a.h[0] = *(const v8h*)base;
    a.h[1] = *(const v8h*)(base + 8);
    return a;
}

// -------- persistent 2-layer LSTM --------
__global__ __launch_bounds__(BLK, 1) void lstm_persistent(
    const float* __restrict__ xg,
    const float* __restrict__ Wih0g, const float* __restrict__ Whh0g,
    const float* __restrict__ bih0g, const float* __restrict__ bhh0g,
    const float* __restrict__ Wih1g, const float* __restrict__ Whh1g,
    const float* __restrict__ bih1g, const float* __restrict__ bhh1g,
    const float* __restrict__ Wling, const float* __restrict__ bling,
    float* __restrict__ outp,
    _Float16* __restrict__ h0buf, _Float16* __restrict__ h1buf,
    float* __restrict__ inpbuf, unsigned* bar)
{
    extern __shared__ char smem[];
    _Float16* w0   = (_Float16*)smem;            // [64][512] f16 : Whh0 slice
    _Float16* w1i  = w0  + 64 * HH;              // [64][512] f16 : Wih1 slice
    _Float16* w1h  = w1i + 64 * HH;              // [64][512] f16 : Whh1 slice
    float* bias0   = (float*)(w1h + 64 * HH);    // [64]
    float* bias1   = bias0 + 64;                 // [64]
    float* wih0s   = bias1 + 64;                 // [64]  (IN==1 column of Wih0)
    float* wlin_s  = wih0s + 64;                 // [512]
    float* outred  = wlin_s + HH;                // [8]

    const int tid  = threadIdx.x;
    const int wg   = blockIdx.x;          // 0..31  -> h columns [wg*16, wg*16+16)
    const int lane = tid & 31;
    const int wv   = tid >> 5;            // 0..15  -> rows [wv*16, wv*16+16)
    const int c0   = wg * 16;
    const int m0   = wv * 16;
    const int n    = lane & 15;           // column within tile (C/D & B layouts)
    const int moff = (lane & 16) ? 8 : 0; // C/D row offset for high half-wave
    const int arow = m0 + (lane & 15);    // A-operand row for this lane

    // ---------- pack weight slices to LDS (f32 -> f16), once ----------
    for (int i = tid; i < 64 * HH; i += BLK) {
        int lr = i >> 9, k = i & (HH - 1);
        int grow = (lr >> 4) * HH + c0 + (lr & 15);   // gate-major global row
        w0 [i] = (_Float16)Whh0g[grow * HH + k];
        w1i[i] = (_Float16)Wih1g[grow * HH + k];
        w1h[i] = (_Float16)Whh1g[grow * HH + k];
    }
    if (tid < 64) {
        int grow = (tid >> 4) * HH + c0 + (tid & 15);
        bias0[tid] = bih0g[grow] + bhh0g[grow];
        bias1[tid] = bih1g[grow] + bhh1g[grow];
        wih0s[tid] = Wih0g[grow];
    }
    for (int j = tid; j < HH; j += BLK) wlin_s[j] = Wling[j];
    __syncthreads();

    const float blin_v = bling[0];
    float c0reg[8] = {0.f,0.f,0.f,0.f,0.f,0.f,0.f,0.f};
    float c1reg[8] = {0.f,0.f,0.f,0.f,0.f,0.f,0.f,0.f};
    const v8f zero8 = {0.f,0.f,0.f,0.f,0.f,0.f,0.f,0.f};
    const int BH = BB * HH;

    // per-lane invariant offsets (folded into literal offsets by the compiler)
    const int aOff = arow * HH + ((lane & 16) ? 16 : 0);   // A base offset, + ks*32
    const int bSel = (lane & 16) ? 16 : 0;                 // B K-half select
    const _Float16* bw0  = w0  + n * HH + bSel;            // gate g -> + g*16*HH
    const _Float16* bw1i = w1i + n * HH + bSel;
    const _Float16* bw1h = w1h + n * HH + bSel;

    for (int t = 0; t < TTOT; ++t) {
        const _Float16* h0p = h0buf + (size_t)(t & 1) * BH;
        _Float16*       h0c = h0buf + (size_t)((t + 1) & 1) * BH;
        const _Float16* h1p = h1buf + (size_t)(t & 1) * BH;
        _Float16*       h1c = h1buf + (size_t)((t + 1) & 1) * BH;

        // ================= layer 0: gates = h0(t-1) @ Whh0^T =================
        v8f acc[4] = {zero8, zero8, zero8, zero8};
#pragma unroll 4
        for (int ks = 0; ks < 16; ++ks) {
            const int k0 = ks * 32;
            AV a = ldA(h0p + aOff + k0);
            __builtin_prefetch(h0p + aOff + k0 + 64, 0, 3);    // global_prefetch_b8
            // load ALL gate B tiles first -> one ds clause, one wait, then 4 WMMAs
            AV b0 = ldB(bw0 + 0 * 16 * HH + k0);
            AV b1 = ldB(bw0 + 1 * 16 * HH + k0);
            AV b2 = ldB(bw0 + 2 * 16 * HH + k0);
            AV b3 = ldB(bw0 + 3 * 16 * HH + k0);
            acc[0] = __builtin_amdgcn_wmma_f32_16x16x32_f16(false, a.v, false, b0.v, (short)0, acc[0], false, false);
            acc[1] = __builtin_amdgcn_wmma_f32_16x16x32_f16(false, a.v, false, b1.v, (short)0, acc[1], false, false);
            acc[2] = __builtin_amdgcn_wmma_f32_16x16x32_f16(false, a.v, false, b2.v, (short)0, acc[2], false, false);
            acc[3] = __builtin_amdgcn_wmma_f32_16x16x32_f16(false, a.v, false, b3.v, (short)0, acc[3], false, false);
        }
        // rank-1 input term + bias + cell update
        float inpv[8];
#pragma unroll
        for (int r = 0; r < 8; ++r) {
            int b = m0 + r + moff;
            inpv[r] = (t < TT_) ? xg[b * TT_ + t] : inpbuf[b];
        }
#pragma unroll
        for (int r = 0; r < 8; ++r) {
            int b = m0 + r + moff;
            float iv = acc[0][r] + bias0[n]      + inpv[r] * wih0s[n];
            float fv = acc[1][r] + bias0[16 + n] + inpv[r] * wih0s[16 + n];
            float gv = acc[2][r] + bias0[32 + n] + inpv[r] * wih0s[32 + n];
            float ov = acc[3][r] + bias0[48 + n] + inpv[r] * wih0s[48 + n];
            float cc = sigf(fv) * c0reg[r] + sigf(iv) * tanhf(gv);
            c0reg[r] = cc;
            float hh = sigf(ov) * tanhf(cc);
            h0c[b * HH + c0 + n] = (_Float16)hh;
        }
        gridbar(bar, tid);   // h0(t) complete, device-wide

        // ======= layer 1: gates = h0(t) @ Wih1^T + h1(t-1) @ Whh1^T =======
        // separate accumulator sets -> 8 independent WMMA dependency chains
        v8f accI[4] = {zero8, zero8, zero8, zero8};
        v8f accH[4] = {zero8, zero8, zero8, zero8};
#pragma unroll 4
        for (int ks = 0; ks < 16; ++ks) {
            const int k0 = ks * 32;
            AV a1 = ldA(h0c + aOff + k0);
            AV a2 = ldA(h1p + aOff + k0);
            __builtin_prefetch(h1p + aOff + k0 + 64, 0, 3);
            AV bi0 = ldB(bw1i + 0 * 16 * HH + k0);
            AV bi1 = ldB(bw1i + 1 * 16 * HH + k0);
            AV bi2 = ldB(bw1i + 2 * 16 * HH + k0);
            AV bi3 = ldB(bw1i + 3 * 16 * HH + k0);
            AV bh0 = ldB(bw1h + 0 * 16 * HH + k0);
            AV bh1 = ldB(bw1h + 1 * 16 * HH + k0);
            AV bh2 = ldB(bw1h + 2 * 16 * HH + k0);
            AV bh3 = ldB(bw1h + 3 * 16 * HH + k0);
            accI[0] = __builtin_amdgcn_wmma_f32_16x16x32_f16(false, a1.v, false, bi0.v, (short)0, accI[0], false, false);
            accI[1] = __builtin_amdgcn_wmma_f32_16x16x32_f16(false, a1.v, false, bi1.v, (short)0, accI[1], false, false);
            accI[2] = __builtin_amdgcn_wmma_f32_16x16x32_f16(false, a1.v, false, bi2.v, (short)0, accI[2], false, false);
            accI[3] = __builtin_amdgcn_wmma_f32_16x16x32_f16(false, a1.v, false, bi3.v, (short)0, accI[3], false, false);
            accH[0] = __builtin_amdgcn_wmma_f32_16x16x32_f16(false, a2.v, false, bh0.v, (short)0, accH[0], false, false);
            accH[1] = __builtin_amdgcn_wmma_f32_16x16x32_f16(false, a2.v, false, bh1.v, (short)0, accH[1], false, false);
            accH[2] = __builtin_amdgcn_wmma_f32_16x16x32_f16(false, a2.v, false, bh2.v, (short)0, accH[2], false, false);
            accH[3] = __builtin_amdgcn_wmma_f32_16x16x32_f16(false, a2.v, false, bh3.v, (short)0, accH[3], false, false);
        }
#pragma unroll
        for (int r = 0; r < 8; ++r) {
            int b = m0 + r + moff;
            float iv = accI[0][r] + accH[0][r] + bias1[n];
            float fv = accI[1][r] + accH[1][r] + bias1[16 + n];
            float gv = accI[2][r] + accH[2][r] + bias1[32 + n];
            float ov = accI[3][r] + accH[3][r] + bias1[48 + n];
            float cc = sigf(fv) * c1reg[r] + sigf(iv) * tanhf(gv);
            c1reg[r] = cc;
            float hh = sigf(ov) * tanhf(cc);
            h1c[b * HH + c0 + n] = (_Float16)hh;
        }
        gridbar(bar, tid);   // h1(t) complete, device-wide

        // ============ output head: out = h1(t) @ Wlin^T + blin ============
        if (tid < 8) outred[tid] = 0.f;
        __syncthreads();
        {
            int b  = wg * 8 + (tid >> 6);   // this WG reduces rows [wg*8, wg*8+8)
            int j0 = (tid & 63) * 8;
            float p = 0.f;
#pragma unroll
            for (int j = 0; j < 8; ++j)
                p += (float)h1c[b * HH + j0 + j] * wlin_s[j0 + j];
            atomicAdd(&outred[tid >> 6], p);   // ds_add_f32
        }
        __syncthreads();
        if (tid < 8) {
            int b = wg * 8 + tid;
            float o = outred[tid] + blin_v;
            outp[b * TTOT + t] = o;
            if (t >= TT_ - 1) inpbuf[b] = o;   // autoregressive feedback
        }
        if (t >= TT_ - 1) gridbar(bar, tid);   // feedback visible before next layer-0
    }
}

extern "C" void kernel_launch(void* const* d_in, const int* in_sizes, int n_in,
                              void* d_out, int out_size, void* d_ws, size_t ws_size,
                              hipStream_t stream) {
    (void)in_sizes; (void)n_in; (void)out_size; (void)ws_size;
    const float* xg    = (const float*)d_in[0];
    const float* Wih0g = (const float*)d_in[1];
    const float* Whh0g = (const float*)d_in[2];
    const float* bih0g = (const float*)d_in[3];
    const float* bhh0g = (const float*)d_in[4];
    const float* Wih1g = (const float*)d_in[5];
    const float* Whh1g = (const float*)d_in[6];
    const float* bih1g = (const float*)d_in[7];
    const float* bhh1g = (const float*)d_in[8];
    const float* Wling = (const float*)d_in[9];
    const float* bling = (const float*)d_in[10];
    float* outp = (float*)d_out;

    // workspace carve-up
    _Float16* h0buf = (_Float16*)d_ws;                 // 2 * B*H f16 (double buffer)
    _Float16* h1buf = h0buf + 2 * BB * HH;             // 2 * B*H f16
    float*    inpbf = (float*)(h1buf + 2 * BB * HH);   // B f32
    unsigned* bar   = (unsigned*)(inpbf + BB);         // {count, generation}

    lstm_init<<<dim3(256), dim3(256), 0, stream>>>(bar, h0buf, h1buf);

    // dynamic LDS: 3 * 64*512 f16 + (64+64+64+512+8) f32 = 199456 B  (< 320 KB/WGP)
    const size_t smem = (size_t)3 * 64 * HH * sizeof(_Float16)
                      + (size_t)(64 + 64 + 64 + HH + 8) * sizeof(float);
    lstm_persistent<<<dim3(NWG), dim3(BLK), smem, stream>>>(
        xg, Wih0g, Whh0g, bih0g, bhh0g, Wih1g, Whh1g, bih1g, bhh1g,
        Wling, bling, outp, h0buf, h1buf, inpbf, bar);
}
